// Masked_Attention_68015102099775
// MI455X (gfx1250) — matrix operand order repaired and checked by reading, not compiled, not run on hardware
//
#include <hip/hip_runtime.h>
#include <hip/hip_bf16.h>
#include <cstdint>
#include <cstddef>

// ---------------------------------------------------------------------------
// CDNA5 (gfx1250) bf16 WMMA attention pipeline, round 4.
//  - BK=64: 16 WMMAs + one barrier per K-step (halved sync overhead).
//  - __launch_bounds__(256,1): let the compiler keep all fragments live.
//  - gfx1250 async global->LDS path (ASYNCcnt) with corrected builtin
//    signature: (int4 __device__*, int4 __shared__*, imm offset, imm cpol).
// ---------------------------------------------------------------------------

typedef __attribute__((ext_vector_type(16))) __bf16 v16bf;
typedef __attribute__((ext_vector_type(8)))  float  v8f;

#define BM 128
#define BN 128
#define BK 64
#define LDSP 72   // BK + 8 bf16 pad -> 144B row stride, conflict-free b128 reads

#if defined(__AMDGCN__) && \
    __has_builtin(__builtin_amdgcn_global_load_async_to_lds_b128) && \
    __has_builtin(__builtin_amdgcn_s_wait_asynccnt)
#define ASYNC_LDS 1
#else
#define ASYNC_LDS 0
#endif

enum GemmMode {
    MODE_PROJ   = 0,  // C bf16 row-major
    MODE_PROJT  = 1,  // C bf16 TRANSPOSED (used to produce V^T)
    MODE_SCORES = 2,  // C f32, causal block skip
    MODE_CTX    = 3,  // C bf16, K-loop bounded at the diagonal
    MODE_OUT    = 4   // C f32 + bias
};

union Frag16 {
    uint4  u[2];
    v16bf  v;
    __bf16 h[16];
};

#if ASYNC_LDS
typedef int i4v __attribute__((vector_size(4 * sizeof(int))));
typedef __attribute__((address_space(1))) i4v* i4v_global_ptr;
typedef __attribute__((address_space(3))) i4v* i4v_shared_ptr;

__device__ __forceinline__ void async_copy16(const __bf16* g, __bf16* l)
{
    __builtin_amdgcn_global_load_async_to_lds_b128(
        (i4v_global_ptr)g, (i4v_shared_ptr)l, 0, 0);
}
#endif

// ---------------------------------------------------------------------------
// Tiled GEMM: C[M,N] = A[M,K] * B^T  where B is stored N-major ([N][K] rows).
// ---------------------------------------------------------------------------
template <int MODE>
__global__ __launch_bounds__(256, 1)
void gemm_wmma(const __bf16* __restrict__ A, const __bf16* __restrict__ B,
               void* __restrict__ Cout, const float* __restrict__ bias,
               int M, int N, int K, int lda, int ldb, int ldc,
               size_t sA, size_t sB, size_t sC)
{
    __shared__ __bf16 As[2][BM][LDSP];   // [buf][m][k]
    __shared__ __bf16 Bs[2][BN][LDSP];   // [buf][n][k]

    const int m0 = blockIdx.y * BM;
    const int n0 = blockIdx.x * BN;

    // Fully-masked causal score blocks: nothing to do.
    if (MODE == MODE_SCORES && n0 > m0 + (BM - 1)) return;

    A += sA * blockIdx.z;
    B += sB * blockIdx.z;

    const int tid  = threadIdx.x;
    const int lane = tid & 31;
    const int wave = tid >> 5;
    const int wm   = wave >> 2;   // 0..1  -> 64-row strip
    const int wn   = wave & 3;    // 0..3  -> 32-col strip

    int kEnd = K;
    if (MODE == MODE_CTX) {       // P is zero beyond the diagonal
        int ke = m0 + BM;
        kEnd = ke < K ? ke : K;
    }
    const int numK = kEnd / BK;

    v8f acc[4][2];
#pragma unroll
    for (int i = 0; i < 4; ++i)
#pragma unroll
        for (int j = 0; j < 2; ++j)
#pragma unroll
            for (int e = 0; e < 8; ++e) acc[i][j][e] = 0.0f;

    // Tile loader: 32 bf16 (4 x b128) per thread for each of A and B.
    const int trow = tid >> 1;            // 0..127
    const int tcol = (tid & 1) * 32;      // 0 or 32
    const __bf16* aPtr = A + (size_t)(m0 + trow) * lda + tcol;
    const __bf16* bPtr = B + (size_t)(n0 + trow) * ldb + tcol;

    const int r15 = lane & 15;
    const int hi  = lane >> 4;

    // ---- preload tile 0 into buffer 0 ----
#if ASYNC_LDS
#pragma unroll
    for (int c = 0; c < 4; ++c) {
        async_copy16(aPtr + c * 8, &As[0][trow][tcol + c * 8]);
        async_copy16(bPtr + c * 8, &Bs[0][trow][tcol + c * 8]);
    }
    __builtin_amdgcn_s_wait_asynccnt(0);
#else
    {
        uint4 ra[4], rb[4];
#pragma unroll
        for (int c = 0; c < 4; ++c) {
            ra[c] = *(const uint4*)(aPtr + c * 8);
            rb[c] = *(const uint4*)(bPtr + c * 8);
        }
#pragma unroll
        for (int c = 0; c < 4; ++c) {
            *(uint4*)&As[0][trow][tcol + c * 8] = ra[c];
            *(uint4*)&Bs[0][trow][tcol + c * 8] = rb[c];
        }
    }
#endif
    __syncthreads();

#if !ASYNC_LDS
    uint4 pa[4], pb[4];
#endif

    for (int kt = 0; kt < numK; ++kt) {
        const int  cur  = kt & 1;
        const int  nxt  = cur ^ 1;
        const bool more = (kt + 1) < numK;

        // Kick off next tile's global->LDS movement before the matrix work.
        if (more) {
            const __bf16* ap = aPtr + (size_t)(kt + 1) * BK;
            const __bf16* bp = bPtr + (size_t)(kt + 1) * BK;
#if ASYNC_LDS
#pragma unroll
            for (int c = 0; c < 4; ++c) {
                async_copy16(ap + c * 8, &As[nxt][trow][tcol + c * 8]);
                async_copy16(bp + c * 8, &Bs[nxt][trow][tcol + c * 8]);
            }
#else
#pragma unroll
            for (int c = 0; c < 4; ++c) {
                pa[c] = *(const uint4*)(ap + c * 8);
                pb[c] = *(const uint4*)(bp + c * 8);
            }
#endif
        }

        // ---- fragments + 16 WMMAs on current buffer (two 32-deep halves) ----
#pragma unroll
        for (int h = 0; h < 2; ++h) {
            Frag16 af[4], bf[2];
#pragma unroll
            for (int i = 0; i < 4; ++i) {
                // A 16x32 frag: lanes 0-15 -> M=lane, K {0..7,16..23};
                //               lanes 16-31 -> M=lane-16, K {8..15,24..31}
                const __bf16* p = &As[cur][wm * 64 + i * 16 + r15][h * 32 + hi * 8];
                af[i].u[0] = *(const uint4*)p;
                af[i].u[1] = *(const uint4*)(p + 16);
            }
#pragma unroll
            for (int j = 0; j < 2; ++j) {
                // B 32x16 frag: lanes 0-15 -> N=lane, K {0..7,16..23};
                //               lanes 16-31 -> N=lane-16, K {8..15,24..31}
                const __bf16* p = &Bs[cur][wn * 32 + j * 16 + r15][h * 32 + hi * 8];
                bf[j].u[0] = *(const uint4*)p;
                bf[j].u[1] = *(const uint4*)(p + 16);
            }
#pragma unroll
            for (int i = 0; i < 4; ++i)
#pragma unroll
                for (int j = 0; j < 2; ++j)
                    acc[i][j] = __builtin_amdgcn_wmma_f32_16x16x32_bf16(
                        false, af[i].v, false, bf[j].v,
                        (short)0, acc[i][j], false, false);
        }

        // Commit / drain next tile into the other buffer.
        if (more) {
#if ASYNC_LDS
            __builtin_amdgcn_s_wait_asynccnt(0);
#else
#pragma unroll
            for (int c = 0; c < 4; ++c) {
                *(uint4*)&As[nxt][trow][tcol + c * 8] = pa[c];
                *(uint4*)&Bs[nxt][trow][tcol + c * 8] = pb[c];
            }
#endif
        }
        __syncthreads();
    }

    // ---- epilogue: C 16x16 f32 layout -> M = v + 8*(lane>>4), N = lane&15
    const int colBase = n0 + wn * 32 + r15;
    const int rowBase = m0 + wm * 64 + hi * 8;

#pragma unroll
    for (int i = 0; i < 4; ++i) {
#pragma unroll
        for (int j = 0; j < 2; ++j) {
            const int col = colBase + j * 16;
            float bv = 0.0f;
            if (MODE == MODE_OUT) bv = bias[col];
#pragma unroll
            for (int e = 0; e < 8; ++e) {
                const int row = rowBase + i * 16 + e;
                if (MODE == MODE_PROJ || MODE == MODE_CTX) {
                    ((__bf16*)Cout)[sC * blockIdx.z + (size_t)row * ldc + col] =
                        (__bf16)acc[i][j][e];
                } else if (MODE == MODE_PROJT) {
                    ((__bf16*)Cout)[(size_t)col * ldc + row] = (__bf16)acc[i][j][e];
                } else {
                    ((float*)Cout)[sC * blockIdx.z + (size_t)row * ldc + col] =
                        acc[i][j][e] + bv;
                }
            }
        }
    }
}

// ---------------------------------------------------------------------------
// Causal softmax over one score row; writes bf16 probabilities in place
// (front half of the f32 row). Row q has q+1 valid keys.
// ---------------------------------------------------------------------------
__global__ __launch_bounds__(256)
void softmax_causal(float* __restrict__ scores, int S, float scale)
{
    __shared__ float red[256];
    const int b   = blockIdx.y;
    const int q   = blockIdx.x;
    const int tid = threadIdx.x;

    float*  row  = scores + ((size_t)b * S + q) * S;
    __bf16* prow = (__bf16*)row;
    const int L = q + 1;

    float m = -3.4e38f;
    for (int k = tid; k < L; k += 256) {
        float v = row[k] * scale;
        m = v > m ? v : m;
    }
    red[tid] = m; __syncthreads();
    for (int off = 128; off > 0; off >>= 1) {
        if (tid < off) { float o = red[tid + off]; if (o > red[tid]) red[tid] = o; }
        __syncthreads();
    }
    m = red[0]; __syncthreads();

    float vals[8];   // S = 2048 -> 8 iters of 256
    float s = 0.0f;
#pragma unroll
    for (int i = 0; i < 8; ++i) {
        const int k = tid + i * 256;
        float e = 0.0f;
        if (k < L) e = __expf(row[k] * scale - m);
        vals[i] = e;
        s += e;
    }
    red[tid] = s; __syncthreads();
    for (int off = 128; off > 0; off >>= 1) {
        if (tid < off) red[tid] += red[tid + off];
        __syncthreads();
    }
    const float inv = 1.0f / red[0];
    __syncthreads();   // all reads of row[] complete before bf16 overwrite

#pragma unroll
    for (int i = 0; i < 8; ++i) {
        const int k = tid + i * 256;
        if (k < S) prow[k] = (__bf16)(vals[i] * inv);
    }
}

// ---------------------------------------------------------------------------
// Tiled transpose + f32->bf16: y[c][r] = (bf16)x[r][c]   (R, C multiples of 32)
// ---------------------------------------------------------------------------
__global__ __launch_bounds__(256)
void transpose_f32_to_bf16(const float* __restrict__ x, __bf16* __restrict__ y,
                           int R, int C)
{
    __shared__ float t[32][33];
    const int r0 = blockIdx.y * 32;
    const int c0 = blockIdx.x * 32;
    const int tx = threadIdx.x & 31;
    const int ty = threadIdx.x >> 5;   // 0..7

#pragma unroll
    for (int i = 0; i < 4; ++i)
        t[ty + i * 8][tx] = x[(size_t)(r0 + ty + i * 8) * C + c0 + tx];
    __syncthreads();
#pragma unroll
    for (int i = 0; i < 4; ++i)
        y[(size_t)(c0 + ty + i * 8) * R + r0 + tx] = (__bf16)t[tx][ty + i * 8];
}

// ---------------------------------------------------------------------------
__global__ __launch_bounds__(256)
void f32_to_bf16(const float* __restrict__ x, __bf16* __restrict__ y, size_t n)
{
    size_t i = (size_t)blockIdx.x * blockDim.x + threadIdx.x;
    const size_t stride = (size_t)gridDim.x * blockDim.x;
    for (; i < n; i += stride) y[i] = (__bf16)x[i];
}

// ---------------------------------------------------------------------------
extern "C" void kernel_launch(void* const* d_in, const int* in_sizes, int n_in,
                              void* d_out, int out_size, void* d_ws, size_t ws_size,
                              hipStream_t stream)
{
    constexpr int Bn = 4, S = 2048, D = 1024;

    const float* inp = (const float*)d_in[0];
    const float* Wq  = (const float*)d_in[1];
    const float* Wk  = (const float*)d_in[2];
    const float* Wv  = (const float*)d_in[3];
    const float* Wo  = (const float*)d_in[4];
    const float* bo  = (const float*)d_in[5];

    char* ws = (char*)d_ws;
    size_t off = 0;
    auto alloc = [&](size_t bytes) {
        char* p = ws + off;
        off += (bytes + 255) & ~(size_t)255;
        return p;
    };

    const size_t nInp = (size_t)Bn * S * D;
    const size_t nW   = (size_t)D * D;

    __bf16* inp_bf = (__bf16*)alloc(nInp * 2);   // reused as ctx later
    __bf16* WqT    = (__bf16*)alloc(nW * 2);     // [N][K] = W^T
    __bf16* WkT    = (__bf16*)alloc(nW * 2);
    __bf16* WvT    = (__bf16*)alloc(nW * 2);
    __bf16* WoT    = (__bf16*)alloc(nW * 2);
    __bf16* Qbf    = (__bf16*)alloc(nInp * 2);   // [B*S][D]
    __bf16* Kbf    = (__bf16*)alloc(nInp * 2);   // [B*S][D]
    __bf16* Vt     = (__bf16*)alloc(nInp * 2);   // [D][B*S]  (transposed V)
    float*  scores = (float*)alloc((size_t)Bn * S * S * 4);   // 64MB, L2-resident

    // 1) conversions (weights transposed to N-major)
    f32_to_bf16<<<dim3(2048), 256, 0, stream>>>(inp, inp_bf, nInp);
    {
        dim3 g(D / 32, D / 32);
        transpose_f32_to_bf16<<<g, 256, 0, stream>>>(Wq, WqT, D, D);
        transpose_f32_to_bf16<<<g, 256, 0, stream>>>(Wk, WkT, D, D);
        transpose_f32_to_bf16<<<g, 256, 0, stream>>>(Wv, WvT, D, D);
        transpose_f32_to_bf16<<<g, 256, 0, stream>>>(Wo, WoT, D, D);
    }

    // 2) Q/K/V projections: [B*S, D] x W^T.  V is written transposed.
    {
        dim3 g(D / BN, (Bn * S) / BM, 1);
        gemm_wmma<MODE_PROJ><<<g, 256, 0, stream>>>(inp_bf, WqT, Qbf, nullptr,
            Bn * S, D, D, D, D, D, 0, 0, 0);
        gemm_wmma<MODE_PROJ><<<g, 256, 0, stream>>>(inp_bf, WkT, Kbf, nullptr,
            Bn * S, D, D, D, D, D, 0, 0, 0);
        gemm_wmma<MODE_PROJT><<<g, 256, 0, stream>>>(inp_bf, WvT, Vt, nullptr,
            Bn * S, D, D, D, D, Bn * S, 0, 0, 0);
    }

    // 3) scores = Q K^T per batch (K matrix is naturally N-major here)
    {
        dim3 g(S / BN, S / BM, Bn);
        gemm_wmma<MODE_SCORES><<<g, 256, 0, stream>>>(Qbf, Kbf, scores, nullptr,
            S, S, D, D, D, S,
            (size_t)S * D, (size_t)S * D, (size_t)S * S);
    }

    // 4) causal softmax (mask first, then /sqrt(D)=32); bf16 P in place.
    softmax_causal<<<dim3(S, Bn), 256, 0, stream>>>(scores, S, 1.0f / 32.0f);

    // 5) ctx = P V.  P rows f32-spaced -> lda = 2S.  V^T is [D][B*S]:
    //    batch stride S elements, ldb = B*S.  ctx reuses inp_bf region.
    __bf16* P   = (__bf16*)scores;
    __bf16* ctx = inp_bf;
    {
        dim3 g(D / BN, S / BM, Bn);
        gemm_wmma<MODE_CTX><<<g, 256, 0, stream>>>(P, Vt, ctx, nullptr,
            S, D, S, 2 * S, Bn * S, D,
            (size_t)2 * S * S, (size_t)S, (size_t)S * D);
    }

    // 6) out = ctx W_o^T-form + b_o  (f32 output)
    {
        dim3 g(D / BN, (Bn * S) / BM, 1);
        gemm_wmma<MODE_OUT><<<g, 256, 0, stream>>>(ctx, WoT, d_out, bo,
            Bn * S, D, D, D, D, D, 0, 0, 0);
    }
}
